// Supernetwork_13529146982711
// MI455X (gfx1250) — matrix operand order repaired and checked by reading, not compiled, not run on hardware
//
#include <hip/hip_runtime.h>
#include <hip/hip_fp16.h>

typedef __attribute__((ext_vector_type(16))) _Float16 v16h;
typedef __attribute__((ext_vector_type(8)))  _Float16 v8h;
typedef __attribute__((ext_vector_type(8)))  float    v8f;

constexpr int   BATCH  = 2048;
constexpr int   NBEAM  = 4;
constexpr int   DEPTH_ = 3;
constexpr int   NEXP_  = 4;
constexpr int   DIM    = 128;
constexpr int   NROW   = BATCH * NBEAM;   // 8192
constexpr int   NCH    = 32;
constexpr float EPS_   = 1e-5f;

// ---------------- conv evaluators (recompute-style, VALID) ----------------

__device__ __forceinline__ float conv1_eval(const float* __restrict__ x,
                                            const float* __restrict__ w,
                                            const float* __restrict__ b,
                                            int n, int c, int y, int xx) {
    const float* xp = x + (size_t)n * 1024;         // [2048,1,32,32]
    const float* wp = w + c * 9;                    // [32,1,3,3]
    float s = b[c];
#pragma unroll
    for (int ky = 0; ky < 3; ++ky)
#pragma unroll
        for (int kx = 0; kx < 3; ++kx)
            s += xp[(y + ky) * 32 + (xx + kx)] * wp[ky * 3 + kx];
    return s;
}

__device__ __forceinline__ float conv2_eval(const float* __restrict__ h1,
                                            const float* __restrict__ w,
                                            const float* __restrict__ b,
                                            int n, int c, int y, int xx) {
    float s = b[c];
    const float* wp = w + (size_t)c * NCH * 9;      // [32,32,3,3]
    for (int ci = 0; ci < NCH; ++ci) {
        const float* xp = h1 + ((size_t)n * NCH + ci) * 225;  // [.,32,15,15]
#pragma unroll
        for (int ky = 0; ky < 3; ++ky)
#pragma unroll
            for (int kx = 0; kx < 3; ++kx)
                s += xp[(y + ky) * 15 + (xx + kx)] * wp[ci * 9 + ky * 3 + kx];
    }
    return s;
}

__device__ __forceinline__ float conv3_eval(const float* __restrict__ h2,
                                            const float* __restrict__ w,
                                            const float* __restrict__ b,
                                            int n, int c, int y, int xx) {
    float s = b[c];
    const float* wp = w + (size_t)c * NCH * 9;
    for (int ci = 0; ci < NCH; ++ci) {
        const float* xp = h2 + ((size_t)n * NCH + ci) * 36;   // [.,32,6,6]
#pragma unroll
        for (int ky = 0; ky < 3; ++ky)
#pragma unroll
            for (int kx = 0; kx < 3; ++kx)
                s += xp[(y + ky) * 6 + (xx + kx)] * wp[ci * 9 + ky * 3 + kx];
    }
    return s;
}

// ---------------- deterministic BN statistics (two stage) ----------------

template <int L>
__global__ void stats_partial_kernel(const float* __restrict__ in,
                                     const float* __restrict__ w,
                                     const float* __restrict__ b,
                                     float* __restrict__ partial) {
    constexpr int OH = (L == 1) ? 30 : (L == 2) ? 13 : 4;
    constexpr int OW = OH;
    const int c = blockIdx.y;
    const int p = blockIdx.x;
    const int npart = gridDim.x;
    const int E = BATCH * OH * OW;
    float s = 0.f, s2 = 0.f;
    for (int i = p * 256 + threadIdx.x; i < E; i += npart * 256) {
        const int n   = i / (OH * OW);
        const int rem = i % (OH * OW);
        float v;
        if constexpr (L == 1) v = conv1_eval(in, w, b, n, c, rem / OW, rem % OW);
        else if constexpr (L == 2) v = conv2_eval(in, w, b, n, c, rem / OW, rem % OW);
        else v = conv3_eval(in, w, b, n, c, rem / OW, rem % OW);
        s += v; s2 += v * v;
    }
    __shared__ float sh[256], sh2[256];
    sh[threadIdx.x] = s; sh2[threadIdx.x] = s2;
    __syncthreads();
    for (int st = 128; st > 0; st >>= 1) {
        if ((int)threadIdx.x < st) {
            sh[threadIdx.x]  += sh[threadIdx.x + st];
            sh2[threadIdx.x] += sh2[threadIdx.x + st];
        }
        __syncthreads();
    }
    if (threadIdx.x == 0) {
        partial[(c * npart + p) * 2 + 0] = sh[0];
        partial[(c * npart + p) * 2 + 1] = sh2[0];
    }
}

__global__ void stats_finalize_kernel(const float* __restrict__ partial, int npart,
                                      float invE, float* __restrict__ mean,
                                      float* __restrict__ invstd) {
    const int c = threadIdx.x;
    if (c >= NCH) return;
    float s = 0.f, s2 = 0.f;
    for (int p = 0; p < npart; ++p) {         // fixed order -> deterministic
        s  += partial[(c * npart + p) * 2 + 0];
        s2 += partial[(c * npart + p) * 2 + 1];
    }
    const float m = s * invE;
    const float v = s2 * invE - m * m;
    mean[c] = m;
    invstd[c] = rsqrtf(v + EPS_);
}

// ---------- fused conv(recompute) + BN + ReLU + 2x2 avg-pool ----------

template <int L>
__global__ void bnpool_kernel(const float* __restrict__ in,
                              const float* __restrict__ w,
                              const float* __restrict__ b,
                              const float* __restrict__ mean,
                              const float* __restrict__ invstd,
                              const float* __restrict__ g,
                              const float* __restrict__ bt,
                              float* __restrict__ out) {
    constexpr int OH = (L == 1) ? 30 : (L == 2) ? 13 : 4;
    constexpr int PH = OH / 2;
    const int tid = blockIdx.x * 256 + threadIdx.x;
    const int total = BATCH * NCH * PH * PH;
    if (tid >= total) return;
    int t = tid;
    const int px = t % PH; t /= PH;
    const int py = t % PH; t /= PH;
    const int c  = t % NCH;
    const int n  = t / NCH;
    const float mu = mean[c], is = invstd[c], gg = g[c], bb = bt[c];
    float acc = 0.f;
#pragma unroll
    for (int dy = 0; dy < 2; ++dy)
#pragma unroll
        for (int dx = 0; dx < 2; ++dx) {
            float v;
            if constexpr (L == 1) v = conv1_eval(in, w, b, n, c, 2 * py + dy, 2 * px + dx);
            else if constexpr (L == 2) v = conv2_eval(in, w, b, n, c, 2 * py + dy, 2 * px + dx);
            else v = conv3_eval(in, w, b, n, c, 2 * py + dy, 2 * px + dx);
            v = (v - mu) * is * gg + bb;
            acc += fmaxf(v, 0.f);
        }
    acc *= 0.25f;
    if constexpr (L < 3)
        out[((size_t)(n * NCH + c) * PH + py) * PH + px] = acc;
    else  // feat = h.reshape(B, 128): index = c*4 + py*2 + px
        out[(size_t)n * DIM + c * 4 + py * 2 + px] = acc;
}

// ---------------- prep: beam broadcast, routing, We transpose+f16 ----------------

__global__ void bcast_kernel(const float* __restrict__ feat,
                             float* __restrict__ hs, _Float16* __restrict__ hs_h) {
    const int tid = blockIdx.x * 256 + threadIdx.x;
    if (tid >= NROW * DIM) return;
    const int r = tid / DIM, d = tid % DIM;
    const float v = feat[(r >> 2) * DIM + d];
    hs[tid]   = v;
    hs_h[tid] = (_Float16)v;
}

__global__ void argmax_kernel(const float* __restrict__ traj, int* __restrict__ idx) {
    const int tid = blockIdx.x * 256 + threadIdx.x;
    if (tid >= DEPTH_ * NROW) return;
    const float* t = traj + (size_t)tid * NEXP_;
    float best = t[0];
    int bi = 0;
#pragma unroll
    for (int e = 1; e < NEXP_; ++e)
        if (t[e] > best) { best = t[e]; bi = e; }   // first max wins (jnp.argmax)
    idx[tid] = bi;
}

// WeT_h[e][n][k] = (f16) We[e][k][n] : makes each lane's B fragment contiguous
__global__ void we_transpose_kernel(const float* __restrict__ We,
                                    _Float16* __restrict__ WeT_h) {
    const int tid = blockIdx.x * 256 + threadIdx.x;
    if (tid >= 3 * DIM * DIM) return;
    const int k = tid % DIM;
    int t = tid / DIM;
    const int n = t % DIM;
    const int e = t / DIM;
    WeT_h[tid] = (_Float16)We[((size_t)e * DIM + k) * DIM + n];
}

// ---------------- MoE step: WMMA f16->f32 GEMM for 3 experts ----------------
// grid = 512 (M-tiles of 16 rows), block = 256 (8 waves = 8 N-tiles of 16 cols)
// All fragments are contiguous 16B/32B vector loads; accumulators stay in VGPRs.

__global__ __launch_bounds__(256)
void moe_step_kernel(const float*    __restrict__ hsin,    // f32 [8192][128] (identity path)
                     const _Float16* __restrict__ hsin_h,  // f16 [8192][128]
                     const _Float16* __restrict__ WeT_h,   // f16 [3][n=128][k=128]
                     const float*    __restrict__ be,      // [3][128]
                     const int*      __restrict__ idx,     // [8192]
                     float*    __restrict__ hsout,
                     _Float16* __restrict__ hsout_h) {
    const int lane = threadIdx.x & 31;
    const int wave = threadIdx.x >> 5;
    const int m0 = blockIdx.x * 16;
    const int n0 = wave * 16;
    const int lo = lane & 15;
    const int hi = lane >> 4;

    v8f acc0 = {}, acc1 = {}, acc2 = {};

    const int mA = m0 + lo;   // A row held by this lane
    const int nB = n0 + lo;   // B column held by this lane
    const _Float16* arow = hsin_h + (size_t)mA * DIM;
    const _Float16* br0  = WeT_h + ((size_t)0 * DIM + nB) * DIM;
    const _Float16* br1  = WeT_h + ((size_t)1 * DIM + nB) * DIM;
    const _Float16* br2  = WeT_h + ((size_t)2 * DIM + nB) * DIM;

#pragma unroll
    for (int kb = 0; kb < 4; ++kb) {
        const int k0 = kb * 32;
        // A 16x32 f16 layout: lane holds K = k0+16*hi+{0..15}, matching B
        const v8h alo = *(const v8h*)(arow + k0 + 16 * hi);
        const v8h ahi = *(const v8h*)(arow + k0 + 16 * hi + 8);
        const v16h a = __builtin_shufflevector(alo, ahi,
            0, 1, 2, 3, 4, 5, 6, 7, 8, 9, 10, 11, 12, 13, 14, 15);
        // B 32x16 f16, same contiguous k order as A: lane holds K = k0+16*hi+{0..15}, contiguous in WeT_h
        const v16h b0 = *(const v16h*)(br0 + k0 + 16 * hi);
        const v16h b1 = *(const v16h*)(br1 + k0 + 16 * hi);
        const v16h b2 = *(const v16h*)(br2 + k0 + 16 * hi);
        acc0 = __builtin_amdgcn_wmma_f32_16x16x32_f16(false, a, false, b0, (short)0, acc0, false, false);
        acc1 = __builtin_amdgcn_wmma_f32_16x16x32_f16(false, a, false, b1, (short)0, acc1, false, false);
        acc2 = __builtin_amdgcn_wmma_f32_16x16x32_f16(false, a, false, b2, (short)0, acc2, false, false);
    }

    // epilogue: per-row expert select (divergence legal after WMMA), bias, ReLU
#pragma unroll
    for (int j = 0; j < 8; ++j) {
        const int m = m0 + j + 8 * hi;       // C/D layout: VGPR j -> M=j (+8 for hi lanes)
        const int e = idx[m];
        float val;
        if (e == 3) {
            val = hsin[(size_t)m * DIM + nB];                 // identity expert
        } else {
            const float lv = (e == 0) ? acc0[j] : (e == 1) ? acc1[j] : acc2[j];
            val = lv + be[e * DIM + nB];
        }
        val = fmaxf(val, 0.f);
        hsout[(size_t)m * DIM + nB]   = val;
        hsout_h[(size_t)m * DIM + nB] = (_Float16)val;
    }
}

// ---------------- output head + scalar scale ----------------

__global__ void head_kernel(const float* __restrict__ hs, const float* __restrict__ Wo,
                            const float* __restrict__ bo, const float* __restrict__ score,
                            float* __restrict__ out) {
    const int tid = blockIdx.x * 256 + threadIdx.x;
    if (tid >= NROW * 10) return;
    const int r = tid / 10, o = tid % 10;
    const float* hr = hs + (size_t)r * DIM;
    const float* wr = Wo + (size_t)o * DIM;
    float s = 0.f;
#pragma unroll 8
    for (int d = 0; d < DIM; ++d) s += hr[d] * wr[d];
    float sc = 1.f;
#pragma unroll
    for (int i = 0; i < NBEAM * DEPTH_; ++i) sc *= score[i];
    out[tid] = (s + bo[o]) * sc;
}

// ---------------- host orchestration ----------------

extern "C" void kernel_launch(void* const* d_in, const int* in_sizes, int n_in,
                              void* d_out, int out_size, void* d_ws, size_t ws_size,
                              hipStream_t stream) {
    const float* x    = (const float*)d_in[0];
    const float* traj = (const float*)d_in[1];
    const float* score= (const float*)d_in[2];
    const float* c1w  = (const float*)d_in[3];
    const float* c1b  = (const float*)d_in[4];
    const float* c2w  = (const float*)d_in[5];
    const float* c2b  = (const float*)d_in[6];
    const float* c3w  = (const float*)d_in[7];
    const float* c3b  = (const float*)d_in[8];
    const float* g1   = (const float*)d_in[9];
    const float* bt1  = (const float*)d_in[10];
    const float* g2   = (const float*)d_in[11];
    const float* bt2  = (const float*)d_in[12];
    const float* g3   = (const float*)d_in[13];
    const float* bt3  = (const float*)d_in[14];
    const float* We   = (const float*)d_in[15];
    const float* be   = (const float*)d_in[16];
    const float* Wo   = (const float*)d_in[17];
    const float* bo   = (const float*)d_in[18];
    float* out = (float*)d_out;

    float* ws = (float*)d_ws;
    size_t off = 0;
    auto alloc = [&](size_t n) -> float* {       // n in floats, 1KB aligned chunks
        float* p = ws + off;
        off += (n + 255) & ~size_t(255);
        return p;
    };
    float*     h1    = alloc((size_t)BATCH * NCH * 15 * 15);   // 59 MB
    float*     h2    = alloc((size_t)BATCH * NCH * 6 * 6);     // 9.4 MB
    float*     feat  = alloc((size_t)BATCH * DIM);             // 1 MB
    float*     hsA   = alloc((size_t)NROW * DIM);              // 4 MB
    float*     hsB   = alloc((size_t)NROW * DIM);              // 4 MB
    _Float16*  hsAh  = (_Float16*)alloc((size_t)NROW * DIM / 2);   // 2 MB f16
    _Float16*  hsBh  = (_Float16*)alloc((size_t)NROW * DIM / 2);   // 2 MB f16
    _Float16*  WeTh  = (_Float16*)alloc((size_t)3 * DIM * DIM / 2); // 96 KB f16
    float*     part  = alloc((size_t)NCH * 96 * 2);
    float* mean1 = alloc(NCH); float* is1 = alloc(NCH);
    float* mean2 = alloc(NCH); float* is2 = alloc(NCH);
    float* mean3 = alloc(NCH); float* is3 = alloc(NCH);
    int*   idx   = (int*)alloc((size_t)DEPTH_ * NROW);
    if (ws_size < off * sizeof(float)) return;

    // layer 1: 2048x32x30x30 -> pool -> 15x15
    stats_partial_kernel<1><<<dim3(96, NCH), 256, 0, stream>>>(x, c1w, c1b, part);
    stats_finalize_kernel<<<1, 32, 0, stream>>>(part, 96, 1.f / (BATCH * 30 * 30), mean1, is1);
    bnpool_kernel<1><<<(BATCH * NCH * 15 * 15) / 256, 256, 0, stream>>>(
        x, c1w, c1b, mean1, is1, g1, bt1, h1);

    // layer 2: -> 13x13 -> pool -> 6x6
    stats_partial_kernel<2><<<dim3(64, NCH), 256, 0, stream>>>(h1, c2w, c2b, part);
    stats_finalize_kernel<<<1, 32, 0, stream>>>(part, 64, 1.f / (BATCH * 13 * 13), mean2, is2);
    bnpool_kernel<2><<<(BATCH * NCH * 6 * 6) / 256, 256, 0, stream>>>(
        h1, c2w, c2b, mean2, is2, g2, bt2, h2);

    // layer 3: -> 4x4 -> pool -> 2x2 => feat [2048,128]
    stats_partial_kernel<3><<<dim3(8, NCH), 256, 0, stream>>>(h2, c3w, c3b, part);
    stats_finalize_kernel<<<1, 32, 0, stream>>>(part, 8, 1.f / (BATCH * 4 * 4), mean3, is3);
    bnpool_kernel<3><<<(BATCH * NCH * 2 * 2) / 256, 256, 0, stream>>>(
        h2, c3w, c3b, mean3, is3, g3, bt3, feat);

    // prep: beam broadcast (f32+f16), routing table, We transpose->f16
    bcast_kernel<<<(NROW * DIM) / 256, 256, 0, stream>>>(feat, hsA, hsAh);
    argmax_kernel<<<(DEPTH_ * NROW) / 256, 256, 0, stream>>>(traj, idx);
    we_transpose_kernel<<<(3 * DIM * DIM) / 256, 256, 0, stream>>>(We, WeTh);

    // 3 MoE steps, WMMA GEMMs, ping-pong hs buffers (f32 + f16 shadows)
    moe_step_kernel<<<NROW / 16, 256, 0, stream>>>(hsA, hsAh, WeTh, be, idx + 0 * NROW, hsB, hsBh);
    moe_step_kernel<<<NROW / 16, 256, 0, stream>>>(hsB, hsBh, WeTh, be, idx + 1 * NROW, hsA, hsAh);
    moe_step_kernel<<<NROW / 16, 256, 0, stream>>>(hsA, hsAh, WeTh, be, idx + 2 * NROW, hsB, hsBh);

    // head + scale
    head_kernel<<<(NROW * 10) / 256, 256, 0, stream>>>(hsB, Wo, bo, score, out);
}
